// PointerGeneratorSeq2Seq_82085414961499
// MI455X (gfx1250) — matrix operand order repaired and checked by reading, not compiled, not run on hardware
//
#include <hip/hip_runtime.h>
#include <hip/hip_bf16.h>
#include <math.h>

#define Bsz 8
#define Ssz 512
#define Tsz 256
#define Dsz 512
#define Vsz 32000
#define D3 (3*Dsz)
#define D2 (2*Dsz)

typedef unsigned short u16;
typedef __attribute__((ext_vector_type(16))) __bf16 v16bf;
typedef __attribute__((ext_vector_type(8)))  float  v8f;
typedef __attribute__((ext_vector_type(4)))  unsigned int u32x4;
typedef __attribute__((ext_vector_type(8)))  int i32x8;
typedef __attribute__((ext_vector_type(4)))  int i32x4;

union Frag { v16bf v; uint4 q[2]; u16 s[16]; };

__device__ __forceinline__ u16 f2bf(float f) {
  unsigned u = __float_as_uint(f);
  u += 0x7FFFu + ((u >> 16) & 1u);
  return (u16)(u >> 16);
}
__device__ __forceinline__ float sigm(float x) { return 1.0f / (1.0f + expf(-x)); }

// Issue a TDM 2D tile load: rows x 32 bf16 elements, row stride K elements,
// from global gptr into LDS at byte offset lds_addr. D# per CDNA5 ISA ch.8.
__device__ __forceinline__ void tdm_load_tile_2d(unsigned lds_addr, const void* gptr,
                                                 unsigned rows, unsigned K) {
  unsigned long long ga = (unsigned long long)(size_t)gptr;
  // group0: [1:0]=count=1, [63:32]=lds_addr, [120:64]=global_addr, [127:126]=type=2
  u32x4 g0 = { 1u,
               lds_addr,
               (unsigned)(ga & 0xFFFFFFFFull),
               (unsigned)((ga >> 32) & 0x01FFFFFFull) | (2u << 30) };
  // group1: [17:16]=data_size(2B), [79:48]=tensor_dim0=K, [111:80]=tensor_dim1=rows,
  //         [127:112]=tile_dim0=32, [143:128]=tile_dim1=rows, [159:144]=tile_dim2=0,
  //         [207:160]=tensor_dim0_stride=K
  i32x8 g1 = { (int)(1u << 16),
               (int)(K << 16),
               (int)((K >> 16) | (rows << 16)),
               (int)(32u << 16),
               (int)rows,
               (int)K,
               0, 0 };
  i32x4 gz4 = { 0, 0, 0, 0 };
  i32x8 gz8 = { 0, 0, 0, 0, 0, 0, 0, 0 };
  __builtin_amdgcn_tensor_load_to_lds(g0, g1, gz4, gz4, gz8, 0);
}

// ---------------------------------------------------------------- utilities
__global__ void k_zero(unsigned char* p, int n) {
  int i = blockIdx.x * blockDim.x + threadIdx.x;
  if (i < n) p[i] = 0;
}

__global__ void k_f2bf(const float* __restrict__ in, u16* __restrict__ out, int n) {
  int i = blockIdx.x * blockDim.x + threadIdx.x;
  if (i < n) out[i] = f2bf(in[i]);
}

// emb[b,s,d] = token_embed[tok[b,s],d] + pos_embed[s,d]  (f32 + bf16 shadow)
__global__ void k_embed(const int* __restrict__ tok, const float* __restrict__ te,
                        const float* __restrict__ pe, float* __restrict__ of,
                        u16* __restrict__ ob, int L) {
  int i = blockIdx.x * blockDim.x + threadIdx.x;   // over Bsz*L*Dsz (exact multiple of 256)
  int d  = i % Dsz;
  int bs = i / Dsz;
  int s  = bs % L;
  float v = te[(size_t)tok[bs] * Dsz + d] + pe[(size_t)s * Dsz + d];
  of[i] = v;
  ob[i] = f2bf(v);
}

// ---------------------------------------------------------------- WMMA GEMM
// C(MxN) = A(MxK) @ W(NxK)^T + bias.  A, W bf16; C f32.
// M,N multiples of 64; K multiple of 32. Block: 256 threads (8 waves), 64x64 tile.
// Tiles are DMA'd into double-buffered LDS by the Tensor Data Mover (wave 0
// issues the descriptors); DMA of tile i+1 overlaps WMMA on tile i.
__global__ __launch_bounds__(256) void k_wmma_gemm(
    const u16* __restrict__ A, const u16* __restrict__ W,
    const float* __restrict__ bias, float* __restrict__ C,
    int M, int N, int K) {
  __shared__ __align__(16) u16 As[2][64][32];
  __shared__ __align__(16) u16 Ws[2][64][32];
  int tid  = threadIdx.x;
  int lane = tid & 31;
  int w    = tid >> 5;         // wave 0..7
  int wm   = w & 3;            // 4 tiles along M
  int wn   = w >> 2;           // 2 groups along N (x2 tiles each)
  int m0   = blockIdx.y * 64;
  int n0   = blockIdx.x * 64;

  int half = lane >> 4;
  int am   = 16 * wm + (lane & 15);
  int ac0  = 16 * half, ac1 = 16 * half + 8;
  int bn0  = 32 * wn + (lane & 15);
  int bn1  = bn0 + 16;
  int bk   = 16 * half;

  const u16* aBase = A + (size_t)m0 * K;
  const u16* wBase = W + (size_t)n0 * K;
  unsigned asAddr[2] = { (unsigned)(size_t)(void*)&As[0][0][0],
                         (unsigned)(size_t)(void*)&As[1][0][0] };
  unsigned wsAddr[2] = { (unsigned)(size_t)(void*)&Ws[0][0][0],
                         (unsigned)(size_t)(void*)&Ws[1][0][0] };

  int nsteps = K >> 5;
  if (w == 0) {
    tdm_load_tile_2d(asAddr[0], aBase, 64u, (unsigned)K);
    tdm_load_tile_2d(wsAddr[0], wBase, 64u, (unsigned)K);
  }

  v8f acc0 = {}; v8f acc1 = {};
  for (int i = 0; i < nsteps; ++i) {
    int buf = i & 1;
    if (w == 0) {
      if (i + 1 < nsteps) {
        const u16* aN = aBase + ((size_t)(i + 1) << 5);
        const u16* wN = wBase + ((size_t)(i + 1) << 5);
        tdm_load_tile_2d(asAddr[buf ^ 1], aN, 64u, (unsigned)K);
        tdm_load_tile_2d(wsAddr[buf ^ 1], wN, 64u, (unsigned)K);
        __builtin_amdgcn_s_wait_tensorcnt(2);   // 2 oldest (current buf) done
      } else {
        __builtin_amdgcn_s_wait_tensorcnt(0);
      }
    }
    __syncthreads();   // publish TDM-written LDS to all waves
    Frag fa, fb0, fb1;
    fa.q[0]  = *(const uint4*)&As[buf][am][ac0];
    fa.q[1]  = *(const uint4*)&As[buf][am][ac1];
    fb0.q[0] = *(const uint4*)&Ws[buf][bn0][bk];
    fb0.q[1] = *(const uint4*)&Ws[buf][bn0][bk + 8];
    fb1.q[0] = *(const uint4*)&Ws[buf][bn1][bk];
    fb1.q[1] = *(const uint4*)&Ws[buf][bn1][bk + 8];
    acc0 = __builtin_amdgcn_wmma_f32_16x16x32_bf16(false, fa.v, false, fb0.v, (short)0, acc0, false, false);
    acc1 = __builtin_amdgcn_wmma_f32_16x16x32_bf16(false, fa.v, false, fb1.v, (short)0, acc1, false, false);
    __syncthreads();   // all waves done with buf before TDM overwrites it
  }

  int cn = n0 + 32 * wn + (lane & 15);
  int cm = m0 + 16 * wm + 8 * half;
  float b0 = bias ? bias[cn] : 0.0f;
  float b1 = bias ? bias[cn + 16] : 0.0f;
#pragma unroll
  for (int r = 0; r < 8; ++r) {
    C[(size_t)(cm + r) * N + cn]      = acc0[r] + b0;
    C[(size_t)(cm + r) * N + cn + 16] = acc1[r] + b1;
  }
}

// ---------------------------------------------------------------- GRU step
// Block j handles d-columns [16j,16j+16). 3 waves -> r/z/n gate segments.
// hp = h(16x512,bf16) @ Whh^T(rows g*512+d..) via WMMA, then fused gate math.
__global__ __launch_bounds__(96) void k_gru_step(
    const float* __restrict__ xp, int t, int L,
    const u16* __restrict__ h_bf, const float* __restrict__ h_f,
    const u16* __restrict__ Whh_bf, const float* __restrict__ bhh,
    float* __restrict__ h_f_out, u16* __restrict__ h_bf_out,
    float* __restrict__ out_seq, u16* __restrict__ out_seq_bf) {
  __shared__ float hp_s[3][16][16];
  int tid  = threadIdx.x;
  int lane = tid & 31;
  int g    = tid >> 5;           // gate segment 0..2
  int d0   = blockIdx.x * 16;
  int half = lane >> 4;
  int m    = lane & 15;
  int nrow = g * Dsz + d0 + m;   // Whh row (weights are (3D, D) row-major)

  v8f acc = {};
  for (int k0 = 0; k0 < Dsz; k0 += 32) {
    Frag fa, fb;
    fa.q[0] = *(const uint4*)&h_bf[(size_t)m * Dsz + k0 + 16 * half];
    fa.q[1] = *(const uint4*)&h_bf[(size_t)m * Dsz + k0 + 16 * half + 8];
    int kk = k0 + 16 * half;
    fb.q[0] = *(const uint4*)&Whh_bf[(size_t)nrow * Dsz + kk];
    fb.q[1] = *(const uint4*)&Whh_bf[(size_t)nrow * Dsz + kk + 8];
    acc = __builtin_amdgcn_wmma_f32_16x16x32_bf16(false, fa.v, false, fb.v, (short)0, acc, false, false);
  }
#pragma unroll
  for (int r = 0; r < 8; ++r)
    hp_s[g][r + 8 * half][lane & 15] = acc[r];
  __syncthreads();

  for (int idx = tid; idx < Bsz * 16; idx += 96) {
    int b = idx >> 4, c = idx & 15;
    int d = d0 + c;
    float hr = hp_s[0][b][c] + bhh[d];
    float hz = hp_s[1][b][c] + bhh[Dsz + d];
    float hn = hp_s[2][b][c] + bhh[2 * Dsz + d];
    const float* xrow = &xp[((size_t)(b * L + t)) * D3];
    float r_ = sigm(xrow[d] + hr);
    float z_ = sigm(xrow[Dsz + d] + hz);
    float n_ = tanhf(xrow[2 * Dsz + d] + r_ * hn);
    float hprev = h_f[(size_t)b * Dsz + d];
    float hnew  = (1.0f - z_) * n_ + z_ * hprev;
    h_f_out[(size_t)b * Dsz + d]  = hnew;
    u16 hb = f2bf(hnew);
    h_bf_out[(size_t)b * Dsz + d] = hb;
    size_t o = ((size_t)(b * L + t)) * Dsz + d;
    out_seq[o]    = hnew;
    out_seq_bf[o] = hb;
  }
}

// ---------------------------------------------------------------- attention
__global__ __launch_bounds__(128) void k_attn_softmax(
    const float* __restrict__ Q, const float* __restrict__ Kmat, float* __restrict__ probs) {
  __shared__ float qs[Dsz];
  __shared__ float red[128];
  int row = blockIdx.x;            // b*T + t
  int b   = row / Tsz;
  int tid = threadIdx.x;
  for (int i = tid; i < Dsz; i += 128) qs[i] = Q[(size_t)row * Dsz + i];
  __syncthreads();
  const float scale = 0.04419417382415922f;  // 1/sqrt(512)
  float v[4];
#pragma unroll
  for (int j = 0; j < 4; ++j) {
    int s = tid + j * 128;
    const float* kr = &Kmat[((size_t)(b * Ssz + s)) * Dsz];
    float acc = 0.f;
    for (int k = 0; k < Dsz; ++k) acc += qs[k] * kr[k];
    v[j] = acc * scale;
  }
  float mx = fmaxf(fmaxf(v[0], v[1]), fmaxf(v[2], v[3]));
  red[tid] = mx; __syncthreads();
  for (int off = 64; off > 0; off >>= 1) { if (tid < off) red[tid] = fmaxf(red[tid], red[tid + off]); __syncthreads(); }
  mx = red[0]; __syncthreads();
  float e[4], ls = 0.f;
#pragma unroll
  for (int j = 0; j < 4; ++j) { e[j] = expf(v[j] - mx); ls += e[j]; }
  red[tid] = ls; __syncthreads();
  for (int off = 64; off > 0; off >>= 1) { if (tid < off) red[tid] += red[tid + off]; __syncthreads(); }
  float inv = 1.0f / red[0];
#pragma unroll
  for (int j = 0; j < 4; ++j) probs[(size_t)row * Ssz + tid + j * 128] = e[j] * inv;
}

// combined = [dec_out, probs @ enc_out]
__global__ __launch_bounds__(128) void k_context_combined(
    const float* __restrict__ probs, const float* __restrict__ enc,
    const float* __restrict__ dec, float* __restrict__ comb) {
  __shared__ float ps[Ssz];
  int row = blockIdx.x;
  int b   = row / Tsz;
  int tid = threadIdx.x;
  for (int i = tid; i < Ssz; i += 128) ps[i] = probs[(size_t)row * Ssz + i];
  __syncthreads();
  for (int d = tid; d < Dsz; d += 128) {
    float acc = 0.f;
    for (int s = 0; s < Ssz; ++s) acc += ps[s] * enc[((size_t)(b * Ssz + s)) * Dsz + d];
    comb[(size_t)row * D2 + Dsz + d] = acc;
    comb[(size_t)row * D2 + d]       = dec[(size_t)row * Dsz + d];
  }
}

__global__ __launch_bounds__(128) void k_gate(
    const float* __restrict__ comb, const float* __restrict__ gW,
    const float* __restrict__ gb, float* __restrict__ gate) {
  __shared__ float red[128];
  int row = blockIdx.x, tid = threadIdx.x;
  float acc = 0.f;
  for (int i = tid; i < D2; i += 128) acc += comb[(size_t)row * D2 + i] * gW[i];
  red[tid] = acc; __syncthreads();
  for (int off = 64; off > 0; off >>= 1) { if (tid < off) red[tid] += red[tid + off]; __syncthreads(); }
  if (tid == 0) gate[row] = sigm(red[0] + gb[0]);
}

// in-place: row = gate * softmax(row) over V
__global__ __launch_bounds__(256) void k_softmax_scale(float* __restrict__ out, const float* __restrict__ gate) {
  __shared__ float red[256];
  int row = blockIdx.x, tid = threadIdx.x;
  float* x = out + (size_t)row * Vsz;
  float mx = -INFINITY;
  for (int i = tid; i < Vsz; i += 256) mx = fmaxf(mx, x[i]);
  red[tid] = mx; __syncthreads();
  for (int off = 128; off > 0; off >>= 1) { if (tid < off) red[tid] = fmaxf(red[tid], red[tid + off]); __syncthreads(); }
  mx = red[0]; __syncthreads();
  float s = 0.f;
  for (int i = tid; i < Vsz; i += 256) s += expf(x[i] - mx);
  red[tid] = s; __syncthreads();
  for (int off = 128; off > 0; off >>= 1) { if (tid < off) red[tid] += red[tid + off]; __syncthreads(); }
  float inv = gate[row] / red[0];
  for (int i = tid; i < Vsz; i += 256) x[i] = expf(x[i] - mx) * inv;
}

// out[b,t,src[b,s]] += (1-gate)*probs ; one thread owns a row -> deterministic
__global__ void k_scatter(float* __restrict__ out, const float* __restrict__ probs,
                          const float* __restrict__ gate, const int* __restrict__ src) {
  int idx = blockIdx.x * blockDim.x + threadIdx.x;
  if (idx >= Bsz * Tsz) return;
  int b = idx / Tsz;
  float wgt = 1.0f - gate[idx];
  float* orow = out + (size_t)idx * Vsz;
  const float* prow = probs + (size_t)idx * Ssz;
  const int* srow = src + b * Ssz;
  for (int s = 0; s < Ssz; ++s) orow[srow[s]] += wgt * prow[s];
}

// ---------------------------------------------------------------- host
static inline size_t alignup(size_t x) { return (x + 255) & ~(size_t)255; }

extern "C" void kernel_launch(void* const* d_in, const int* in_sizes, int n_in,
                              void* d_out, int out_size, void* d_ws, size_t ws_size,
                              hipStream_t stream) {
  (void)in_sizes; (void)n_in; (void)out_size; (void)ws_size;
  const int*   src        = (const int*)d_in[0];
  const int*   tgt        = (const int*)d_in[1];
  const float* token_emb  = (const float*)d_in[2];
  const float* pos_emb    = (const float*)d_in[3];
  const float* enc_Wih    = (const float*)d_in[4];
  const float* enc_Whh    = (const float*)d_in[5];
  const float* enc_bih    = (const float*)d_in[6];
  const float* enc_bhh    = (const float*)d_in[7];
  const float* dec_Wih    = (const float*)d_in[8];
  const float* dec_Whh    = (const float*)d_in[9];
  const float* dec_bih    = (const float*)d_in[10];
  const float* dec_bhh    = (const float*)d_in[11];
  const float* Wq         = (const float*)d_in[12];
  const float* Wk         = (const float*)d_in[13];
  const float* gen_W      = (const float*)d_in[14];
  const float* gen_b      = (const float*)d_in[15];
  const float* gate_W     = (const float*)d_in[16];
  const float* gate_b     = (const float*)d_in[17];
  float* out = (float*)d_out;

  char* p = (char*)d_ws;
  auto take = [&](size_t bytes) -> char* { char* r = p; p += alignup(bytes); return r; };

  float* src_emb    = (float*)take((size_t)Bsz * Ssz * Dsz * 4);
  u16*   src_emb_bf = (u16*)  take((size_t)Bsz * Ssz * Dsz * 2);
  float* tgt_emb    = (float*)take((size_t)Bsz * Tsz * Dsz * 4);
  u16*   tgt_emb_bf = (u16*)  take((size_t)Bsz * Tsz * Dsz * 2);
  float* xp_enc     = (float*)take((size_t)Bsz * Ssz * D3 * 4);
  float* xp_dec     = (float*)take((size_t)Bsz * Tsz * D3 * 4);
  u16*   encWih_bf  = (u16*)  take((size_t)D3 * Dsz * 2);
  u16*   decWih_bf  = (u16*)  take((size_t)D3 * Dsz * 2);
  u16*   encWhh_bf  = (u16*)  take((size_t)D3 * Dsz * 2);
  u16*   decWhh_bf  = (u16*)  take((size_t)D3 * Dsz * 2);
  u16*   Wq_bf      = (u16*)  take((size_t)Dsz * Dsz * 2);
  u16*   Wk_bf      = (u16*)  take((size_t)Dsz * Dsz * 2);
  u16*   genW_bf    = (u16*)  take((size_t)Vsz * D2 * 2);
  char*  hzone      = take(2 * 16 * Dsz * 4 + 2 * 16 * Dsz * 2);
  float* hf0 = (float*)hzone;
  float* hf1 = hf0 + 16 * Dsz;
  u16*   hb0 = (u16*)(hf1 + 16 * Dsz);
  u16*   hb1 = hb0 + 16 * Dsz;
  int    hzone_bytes = 2 * 16 * Dsz * 4 + 2 * 16 * Dsz * 2;
  float* enc_out    = (float*)take((size_t)Bsz * Ssz * Dsz * 4);
  u16*   enc_out_bf = (u16*)  take((size_t)Bsz * Ssz * Dsz * 2);
  float* dec_out    = (float*)take((size_t)Bsz * Tsz * Dsz * 4);
  u16*   dec_out_bf = (u16*)  take((size_t)Bsz * Tsz * Dsz * 2);
  float* Qb         = (float*)take((size_t)Bsz * Tsz * Dsz * 4);
  float* Kb         = (float*)take((size_t)Bsz * Ssz * Dsz * 4);
  float* probs      = (float*)take((size_t)Bsz * Tsz * Ssz * 4);
  float* comb       = (float*)take((size_t)Bsz * Tsz * D2 * 4);
  u16*   comb_bf    = (u16*)  take((size_t)Bsz * Tsz * D2 * 2);
  float* gate       = (float*)take((size_t)Bsz * Tsz * 4);

  auto conv = [&](const float* in, u16* o, int n) {
    k_f2bf<<<(n + 255) / 256, 256, 0, stream>>>(in, o, n);
  };
  conv(enc_Wih, encWih_bf, D3 * Dsz);
  conv(dec_Wih, decWih_bf, D3 * Dsz);
  conv(enc_Whh, encWhh_bf, D3 * Dsz);
  conv(dec_Whh, decWhh_bf, D3 * Dsz);
  conv(Wq, Wq_bf, Dsz * Dsz);
  conv(Wk, Wk_bf, Dsz * Dsz);
  conv(gen_W, genW_bf, Vsz * D2);

  k_embed<<<(Bsz * Ssz * Dsz) / 256, 256, 0, stream>>>(src, token_emb, pos_emb, src_emb, src_emb_bf, Ssz);
  k_embed<<<(Bsz * Tsz * Dsz) / 256, 256, 0, stream>>>(tgt, token_emb, pos_emb, tgt_emb, tgt_emb_bf, Tsz);

  // input projections: xp = emb @ Wih^T + bih
  k_wmma_gemm<<<dim3(D3 / 64, (Bsz * Ssz) / 64), 256, 0, stream>>>(src_emb_bf, encWih_bf, enc_bih, xp_enc, Bsz * Ssz, D3, Dsz);
  k_wmma_gemm<<<dim3(D3 / 64, (Bsz * Tsz) / 64), 256, 0, stream>>>(tgt_emb_bf, decWih_bf, dec_bih, xp_dec, Bsz * Tsz, D3, Dsz);

  // encoder recurrence
  k_zero<<<(hzone_bytes + 255) / 256, 256, 0, stream>>>((unsigned char*)hzone, hzone_bytes);
  {
    float* hf[2] = {hf0, hf1};
    u16*   hb[2] = {hb0, hb1};
    int cur = 0;
    for (int t = 0; t < Ssz; ++t) {
      k_gru_step<<<Dsz / 16, 96, 0, stream>>>(xp_enc, t, Ssz, hb[cur], hf[cur], encWhh_bf, enc_bhh,
                                              hf[1 - cur], hb[1 - cur], enc_out, enc_out_bf);
      cur ^= 1;
    }
  }
  // decoder recurrence
  k_zero<<<(hzone_bytes + 255) / 256, 256, 0, stream>>>((unsigned char*)hzone, hzone_bytes);
  {
    float* hf[2] = {hf0, hf1};
    u16*   hb[2] = {hb0, hb1};
    int cur = 0;
    for (int t = 0; t < Tsz; ++t) {
      k_gru_step<<<Dsz / 16, 96, 0, stream>>>(xp_dec, t, Tsz, hb[cur], hf[cur], decWhh_bf, dec_bhh,
                                              hf[1 - cur], hb[1 - cur], dec_out, dec_out_bf);
      cur ^= 1;
    }
  }

  // attention projections
  k_wmma_gemm<<<dim3(Dsz / 64, (Bsz * Tsz) / 64), 256, 0, stream>>>(dec_out_bf, Wq_bf, nullptr, Qb, Bsz * Tsz, Dsz, Dsz);
  k_wmma_gemm<<<dim3(Dsz / 64, (Bsz * Ssz) / 64), 256, 0, stream>>>(enc_out_bf, Wk_bf, nullptr, Kb, Bsz * Ssz, Dsz, Dsz);

  k_attn_softmax<<<Bsz * Tsz, 128, 0, stream>>>(Qb, Kb, probs);
  k_context_combined<<<Bsz * Tsz, 128, 0, stream>>>(probs, enc_out, dec_out, comb);
  conv(comb, comb_bf, Bsz * Tsz * D2);
  k_gate<<<Bsz * Tsz, 128, 0, stream>>>(comb, gate_W, gate_b, gate);

  // big generator GEMM straight into d_out (logits), then fused softmax*gate + copy scatter
  k_wmma_gemm<<<dim3(Vsz / 64, (Bsz * Tsz) / 64), 256, 0, stream>>>(comb_bf, genW_bf, gen_b, out, Bsz * Tsz, Vsz, D2);
  k_softmax_scale<<<Bsz * Tsz, 256, 0, stream>>>(out, gate);
  k_scatter<<<(Bsz * Tsz + 63) / 64, 64, 0, stream>>>(out, probs, gate, src);
}
